// WideLSTM_20280835572371
// MI455X (gfx1250) — matrix operand order repaired and checked by reading, not compiled, not run on hardware
//
#include <hip/hip_runtime.h>
#include <stdint.h>

typedef __attribute__((ext_vector_type(16))) _Float16 v16h;
typedef __attribute__((ext_vector_type(8)))  _Float16 v8h;
typedef __attribute__((ext_vector_type(4)))  _Float16 v4h;
typedef __attribute__((ext_vector_type(8)))  float    v8f;
typedef __attribute__((ext_vector_type(4)))  float    v4f;

#define LSTM_N 16
#define DIM_I  128
#define DIM_H  128
#define GATES  512   // 4*H
#define BATCH  32
#define SEQ_T  512
#define ROWS   16    // batch rows handled per block (BATCH/2)
#define WPAD   136   // padded LDS row stride in halfs (128 + 8 -> bank-friendly)

__device__ __forceinline__ float fast_sigmoid(float x) {
    return 1.0f / (1.0f + __expf(-x));
}
__device__ __forceinline__ float fast_tanh(float x) {
    x = fminf(fmaxf(x, -15.0f), 15.0f);
    float e = __expf(-2.0f * x);
    return (1.0f - e) / (1.0f + e);
}

__global__ __launch_bounds__(256, 1)
void widelstm_kernel(const float* __restrict__ x,      // [B, T, N*I]
                     const float* __restrict__ W_ih,   // [N, 4H, I]
                     const float* __restrict__ W_hh,   // [N, 4H, H]
                     const float* __restrict__ b_ih,   // [N, 4H]
                     const float* __restrict__ b_hh,   // [N, 4H]
                     float* __restrict__ out)          // output | h_n | c_n (concat)
{
    // LDS-resident f16 weights (the CDNA5 320KB-LDS play)
    __shared__ _Float16 sWih[GATES * WPAD];   // 139,264 B
    __shared__ _Float16 sWhh[GATES * WPAD];   // 139,264 B
    __shared__ _Float16 sX[ROWS * WPAD];      //   4,352 B  (x_t as f16, A-operand)
    __shared__ _Float16 sH[ROWS * WPAD];      //   4,352 B  (h_{t-1} as f16, A-operand)
    __shared__ float    sStage[ROWS * DIM_I]; //   8,192 B  (async-load staging, fp32)
    __shared__ float    sBias[GATES];         //   2,048 B  (b_ih + b_hh)

    const int tid  = threadIdx.x;
    const int lane = tid & 31;
    const int wv   = tid >> 5;              // wave 0..7
    const int n    = blockIdx.x & (LSTM_N - 1);
    const int bh   = blockIdx.x >> 4;       // batch half 0/1

    // ---- Stage weights fp32 -> f16 into padded-row LDS ----
    {
        const float* gih = W_ih + (size_t)n * GATES * DIM_I;
        const float* ghh = W_hh + (size_t)n * GATES * DIM_I;
        #pragma unroll 4
        for (int it = 0; it < (GATES * DIM_I) / (256 * 4); ++it) {
            int flat = (it * 256 + tid) * 4;
            int r = flat >> 7, c = flat & 127;
            v4f a = *(const v4f*)(gih + flat);
            v4f b = *(const v4f*)(ghh + flat);
            v4h ha = { (_Float16)a.x, (_Float16)a.y, (_Float16)a.z, (_Float16)a.w };
            v4h hb = { (_Float16)b.x, (_Float16)b.y, (_Float16)b.z, (_Float16)b.w };
            *(v4h*)(sWih + r * WPAD + c) = ha;
            *(v4h*)(sWhh + r * WPAD + c) = hb;
        }
        for (int i = tid; i < GATES; i += 256)
            sBias[i] = b_ih[n * GATES + i] + b_hh[n * GATES + i];
        for (int i = tid; i < ROWS * WPAD; i += 256)
            sH[i] = (_Float16)0.0f;
    }

    // Async x-tile fetch: wave wv pulls rows 2wv, 2wv+1; each instr = 512B row slice.
    auto issue_x = [&](int t) {
        #pragma unroll
        for (int j = 0; j < 2; ++j) {
            int r = wv * 2 + j;
            const float* g = x + ((size_t)(bh * ROWS + r) * SEQ_T + t) * (size_t)(LSTM_N * DIM_I)
                               + n * DIM_I + lane * 4;
            unsigned ldsoff = (unsigned)(uintptr_t)(sStage + r * DIM_I + lane * 4);
            asm volatile("global_load_async_to_lds_b128 %0, %1, off"
                         :: "v"(ldsoff), "v"(g) : "memory");
        }
    };

    v8f c_reg = {0.f,0.f,0.f,0.f,0.f,0.f,0.f,0.f};
    v8f h_reg = {0.f,0.f,0.f,0.f,0.f,0.f,0.f,0.f};

    issue_x(0);

    const int col  = wv * 16 + (lane & 15);   // hidden column owned in C/D tiles
    const int arow = (lane & 15);             // A-operand M row this lane loads
    const int ac0  = (lane < 16) ? 0 : 8;     // 16-bit A layout K base (ISA 7.12.2)
    const int bc0  = (lane < 16) ? 0 : 8;     // 16-bit B layout K base (same as A)

    float* hnp = out + (size_t)BATCH * SEQ_T * (LSTM_N * DIM_H);
    float* cnp = hnp + (size_t)BATCH * (LSTM_N * DIM_H);

    for (int t = 0; t < SEQ_T; ++t) {
        asm volatile("s_wait_asynccnt 0x0" ::: "memory");
        __syncthreads();                     // staged x_t visible to all waves

        // fp32 stage -> f16 A-buffer (8 elems/thread)
        {
            int flat = tid * 8;
            int r = flat >> 7, c = flat & 127;
            v4f a0 = *(const v4f*)(sStage + flat);
            v4f a1 = *(const v4f*)(sStage + flat + 4);
            v8h hx = { (_Float16)a0.x, (_Float16)a0.y, (_Float16)a0.z, (_Float16)a0.w,
                       (_Float16)a1.x, (_Float16)a1.y, (_Float16)a1.z, (_Float16)a1.w };
            *(v8h*)(sX + r * WPAD + c) = hx;
        }
        __syncthreads();                     // sX ready; stage free for t+1

        if (t + 1 < SEQ_T) issue_x(t + 1);   // overlap next fetch with WMMA work

        // z = [x_t | h_{t-1}] * [W_ih | W_hh]^T + bias   (K = 256, 32 WMMAs/wave)
        v8f acc[4];
        #pragma unroll
        for (int q = 0; q < 4; ++q) {
            float bv = sBias[q * 128 + col];
            #pragma unroll
            for (int r8 = 0; r8 < 8; ++r8) acc[q][r8] = bv;
        }
        #pragma unroll
        for (int ks = 0; ks < 8; ++ks) {
            const _Float16* asrc = (ks < 4) ? sX   : sH;
            const _Float16* wsrc = (ks < 4) ? sWih : sWhh;
            const int kb = (ks & 3) * 32;
            v8h alo = *(const v8h*)(asrc + arow * WPAD + kb + ac0);
            v8h ahi = *(const v8h*)(asrc + arow * WPAD + kb + ac0 + 16);
            v16h A = __builtin_shufflevector(alo, ahi,
                      0,1,2,3,4,5,6,7,8,9,10,11,12,13,14,15);
            #pragma unroll
            for (int q = 0; q < 4; ++q) {
                const int grow = (q * 8 + wv) * 16 + (lane & 15);
                const _Float16* bp = wsrc + grow * WPAD + kb + bc0;
                v8h blo = *(const v8h*)(bp);
                v8h bhi = *(const v8h*)(bp + 16);
                v16h Bm = __builtin_shufflevector(blo, bhi,
                          0,1,2,3,4,5,6,7,8,9,10,11,12,13,14,15);
                acc[q] = __builtin_amdgcn_wmma_f32_16x16x32_f16(
                             false, A, false, Bm, (short)0, acc[q], false, false);
            }
        }
        __syncthreads();                     // all reads of sH/sX complete

        // gates (i,f,g,o), cell update — wave-local, c stays in VGPRs
        #pragma unroll
        for (int r8 = 0; r8 < 8; ++r8) {
            float ig = fast_sigmoid(acc[0][r8]);
            float fg = fast_sigmoid(acc[1][r8]);
            float gg = fast_tanh  (acc[2][r8]);
            float og = fast_sigmoid(acc[3][r8]);
            float cc = fg * c_reg[r8] + ig * gg;
            c_reg[r8] = cc;
            h_reg[r8] = og * fast_tanh(cc);
        }

        // h -> LDS (f16, next step's A) and global output (fp32, coalesced rows)
        #pragma unroll
        for (int r8 = 0; r8 < 8; ++r8) {
            int m = r8 + ((lane >> 4) << 3);             // D-layout M row
            sH[m * WPAD + col] = (_Float16)h_reg[r8];
            out[((size_t)(bh * ROWS + m) * SEQ_T + t) * (size_t)(LSTM_N * DIM_H)
                + n * DIM_H + col] = h_reg[r8];
        }
        __syncthreads();                     // new h visible before next step
    }

    // h_n / c_n
    #pragma unroll
    for (int r8 = 0; r8 < 8; ++r8) {
        int m = r8 + ((lane >> 4) << 3);
        size_t o = (size_t)(bh * ROWS + m) * (LSTM_N * DIM_H) + n * DIM_H + col;
        hnp[o] = h_reg[r8];
        cnp[o] = c_reg[r8];
    }
}

extern "C" void kernel_launch(void* const* d_in, const int* in_sizes, int n_in,
                              void* d_out, int out_size, void* d_ws, size_t ws_size,
                              hipStream_t stream) {
    (void)in_sizes; (void)n_in; (void)out_size; (void)d_ws; (void)ws_size;
    const float* x    = (const float*)d_in[0];
    const float* W_ih = (const float*)d_in[1];
    const float* W_hh = (const float*)d_in[2];
    const float* b_ih = (const float*)d_in[3];
    const float* b_hh = (const float*)d_in[4];
    float* out = (float*)d_out;
    widelstm_kernel<<<dim3(LSTM_N * 2), dim3(256), 0, stream>>>(
        x, W_ih, W_hh, b_ih, b_hh, out);
}
